// MentionScore_23364622090797
// MI455X (gfx1250) — matrix operand order repaired and checked by reading, not compiled, not run on hardware
//
#include <hip/hip_runtime.h>
#include <hip/hip_bf16.h>

// ---------------------------------------------------------------------------
// MentionScore for MI455X (gfx1250, wave32, WMMA f16->f32)
//
// Pipeline:
//   1) states f32 -> f16                              (k_cvt)
//   2) transpose+pad weights to f16 [160 x Kp]        (k_prep_weights)
//   3) H1 = ReLU(statesH @ Wa1 + ba1)   WMMA GEMM     (k_gemm_relu, Kp=1024)
//   4) H2 = ReLU(H1 @ Wa2 + ba2)        WMMA GEMM     (k_gemm_relu, Kp=160)
//   5) attns = H2 @ Wa3 + ba3                          (k_dot150)
//   6) per-span softmax pooling -> pooledH f16,
//      plus padded width-embedding row -> sizeH f16    (k_pool)
//   7) S1 = ReLU(G @ Ws1 + bs1) with G gathered on the
//      fly from statesH/pooledH/sizeH                  (k_gemm_mention)
//   8) S2 = ReLU(S1 @ Ws2 + bs2)        WMMA GEMM      (k_gemm_relu, Kp=160)
//   9) scores = S2 @ Ws3 + bs3 -> d_out                (k_dot150)
//
// sched_barrier(0) between each 5-tile B-load group and its 5 WMMAs forces
// the loads to stay grouped (one clause, one wait, 5 back-to-back WMMAs) and
// lets the scheduler hoist the NEXT group's loads over the current group's
// WMMAs (software pipelining).
// ---------------------------------------------------------------------------

#if defined(__has_builtin)
#  if __has_builtin(__builtin_amdgcn_sched_barrier)
#    define SCHED_FENCE() __builtin_amdgcn_sched_barrier(0)
#  else
#    define SCHED_FENCE() ((void)0)
#  endif
#else
#  define SCHED_FENCE() ((void)0)
#endif

typedef __attribute__((ext_vector_type(16))) _Float16 v16h;
typedef __attribute__((ext_vector_type(8)))  _Float16 h8;
typedef __attribute__((ext_vector_type(4)))  _Float16 h4;
typedef __attribute__((ext_vector_type(8)))  float    v8f;
typedef __attribute__((ext_vector_type(4)))  float    f4;

#define T_TOK 8192
#define A_DIM 1024
#define E_DIM 512
#define S_SPN 32768
#define NPAD  160     // HID=150 padded to 160
#define KP3   2592    // GI=2580 padded to 2592 (last 32-block comes from sizeH)

// Build 16-f16 fragment from two contiguous 8-f16 halves.
__device__ __forceinline__ v16h pack2(h8 lo, h8 hi) {
  return __builtin_shufflevector(lo, hi, 0,1,2,3,4,5,6,7,8,9,10,11,12,13,14,15);
}

// -------------------------- conversion kernels -----------------------------

__global__ void k_cvt(const float* __restrict__ x, _Float16* __restrict__ y, int n) {
  for (int i = blockIdx.x * blockDim.x + threadIdx.x; i < n; i += gridDim.x * blockDim.x)
    y[i] = (_Float16)x[i];
}

// Transpose f32 [K x 150] weights into padded f16 [160 x Kp] (row n = output col).
__global__ void k_prep_weights(const float* __restrict__ Wa1, const float* __restrict__ Wa2,
                               const float* __restrict__ Ws1, const float* __restrict__ Ws2,
                               _Float16* __restrict__ Wa1T, _Float16* __restrict__ Wa2T,
                               _Float16* __restrict__ Ws1T, _Float16* __restrict__ Ws2T) {
  const int n0 = NPAD * 1024;   // Wa1T
  const int n1 = NPAD * NPAD;   // Wa2T
  const int n2 = NPAD * KP3;    // Ws1T
  const int n3 = NPAD * NPAD;   // Ws2T
  const int total = n0 + n1 + n2 + n3;
  for (int i = blockIdx.x * blockDim.x + threadIdx.x; i < total; i += gridDim.x * blockDim.x) {
    int j = i;
    if (j < n0) {
      int n = j / 1024, k = j % 1024;
      Wa1T[j] = (_Float16)((n < 150) ? Wa1[k * 150 + n] : 0.0f);
      continue;
    }
    j -= n0;
    if (j < n1) {
      int n = j / NPAD, k = j % NPAD;
      Wa2T[j] = (_Float16)((n < 150 && k < 150) ? Wa2[k * 150 + n] : 0.0f);
      continue;
    }
    j -= n1;
    if (j < n2) {
      int n = j / KP3, k = j % KP3;
      Ws1T[j] = (_Float16)((n < 150 && k < 2580) ? Ws1[k * 150 + n] : 0.0f);
      continue;
    }
    j -= n2;
    {
      int n = j / NPAD, k = j % NPAD;
      Ws2T[j] = (_Float16)((n < 150 && k < 150) ? Ws2[k * 150 + n] : 0.0f);
    }
  }
}

// ----------------------------- generic GEMM --------------------------------
// Y[M x 160] = ReLU(X[M x Kp] @ W + bias), W given as WT f16 [160 x Kp].
// One wave per 16-row block; 10 N-tiles of 16 in v8f accumulators.
__global__ void __launch_bounds__(128)
k_gemm_relu(const _Float16* __restrict__ X, int Kp,
            const _Float16* __restrict__ WT,
            const float* __restrict__ bias,
            _Float16* __restrict__ Y) {
  const int lane = threadIdx.x & 31;
  const int wid  = threadIdx.x >> 5;
  const int m0   = (blockIdx.x * 4 + wid) * 16;
  const int half = lane >> 4;          // 0 or 1
  const int l16  = lane & 15;
  const int kbA  = half * 8;           // A frag: lanes 0-15 -> K 0..7 & 16..23
  const int kbB  = half * 8;           // B frag: lanes 0-15 -> K 0..7 & 16..23

  const _Float16* arow = X  + (size_t)(m0 + l16) * Kp + kbA;
  const _Float16* brow = WT + (size_t)l16 * Kp + kbB;

  v8f acc[10];
#pragma unroll
  for (int t = 0; t < 10; ++t)
#pragma unroll
    for (int j = 0; j < 8; ++j) acc[t][j] = 0.0f;

  for (int k0 = 0; k0 < Kp; k0 += 32) {
    v16h a = pack2(*(const h8*)(arow + k0), *(const h8*)(arow + k0 + 16));
#pragma unroll
    for (int g = 0; g < 2; ++g) {
      v16h b0, b1, b2, b3, b4;
      {
        const _Float16* bp0 = brow + (size_t)(g * 5 + 0) * 16 * Kp + k0;
        const _Float16* bp1 = brow + (size_t)(g * 5 + 1) * 16 * Kp + k0;
        const _Float16* bp2 = brow + (size_t)(g * 5 + 2) * 16 * Kp + k0;
        const _Float16* bp3 = brow + (size_t)(g * 5 + 3) * 16 * Kp + k0;
        const _Float16* bp4 = brow + (size_t)(g * 5 + 4) * 16 * Kp + k0;
        b0 = pack2(*(const h8*)bp0, *(const h8*)(bp0 + 16));
        b1 = pack2(*(const h8*)bp1, *(const h8*)(bp1 + 16));
        b2 = pack2(*(const h8*)bp2, *(const h8*)(bp2 + 16));
        b3 = pack2(*(const h8*)bp3, *(const h8*)(bp3 + 16));
        b4 = pack2(*(const h8*)bp4, *(const h8*)(bp4 + 16));
      }
      SCHED_FENCE();  // keep the 10 loads grouped ahead of the 5 WMMAs
      acc[g * 5 + 0] = __builtin_amdgcn_wmma_f32_16x16x32_f16(
          false, a, false, b0, (short)0, acc[g * 5 + 0], false, false);
      acc[g * 5 + 1] = __builtin_amdgcn_wmma_f32_16x16x32_f16(
          false, a, false, b1, (short)0, acc[g * 5 + 1], false, false);
      acc[g * 5 + 2] = __builtin_amdgcn_wmma_f32_16x16x32_f16(
          false, a, false, b2, (short)0, acc[g * 5 + 2], false, false);
      acc[g * 5 + 3] = __builtin_amdgcn_wmma_f32_16x16x32_f16(
          false, a, false, b3, (short)0, acc[g * 5 + 3], false, false);
      acc[g * 5 + 4] = __builtin_amdgcn_wmma_f32_16x16x32_f16(
          false, a, false, b4, (short)0, acc[g * 5 + 4], false, false);
    }
  }

  const int rowoff = half * 8;
#pragma unroll
  for (int t = 0; t < 10; ++t) {
    const int n = t * 16 + l16;
    const float bv = (n < 150) ? bias[n] : 0.0f;
#pragma unroll
    for (int j = 0; j < 8; ++j) {
      float v = acc[t][j] + bv;
      v = v > 0.0f ? v : 0.0f;
      Y[(size_t)(m0 + rowoff + j) * NPAD + n] = (_Float16)v;
    }
  }
}

// ---------------------- fused gather + mention GEMM ------------------------
// S1[S x 160] = ReLU(G @ Ws1 + bs1), G[s] = [statesH[start] | statesH[end] |
// pooledH[s] | sizeH[s](32, zero-padded)] gathered per K-block (segment
// boundaries 1024 / 2048 / 2560 are 32-aligned, so each K-block has one source).
__global__ void __launch_bounds__(128)
k_gemm_mention(const _Float16* __restrict__ statesH,
               const _Float16* __restrict__ pooledH,
               const _Float16* __restrict__ sizeH,   // [S x 32] padded width emb
               const int* __restrict__ starts,
               const int* __restrict__ lens,
               const _Float16* __restrict__ WT,      // Ws1T [160 x 2592]
               const float* __restrict__ bias,       // bs1
               _Float16* __restrict__ Y) {           // S1 [S x 160]
  const int lane = threadIdx.x & 31;
  const int wid  = threadIdx.x >> 5;
  const int m0   = (blockIdx.x * 4 + wid) * 16;
  const int half = lane >> 4;
  const int l16  = lane & 15;
  const int kbA  = half * 8;
  const int kbB  = half * 8;

  const int r     = m0 + l16;          // span index for this lane's A row
  const int start = starts[r];
  const int end   = start + lens[r];

  const _Float16* a0 = statesH + (size_t)start * A_DIM + kbA;
  const _Float16* a1 = statesH + (size_t)end   * A_DIM + kbA;
  const _Float16* a2 = pooledH + (size_t)r     * E_DIM + kbA;
  const _Float16* a3 = sizeH   + (size_t)r     * 32    + kbA;
  const _Float16* brow = WT + (size_t)l16 * KP3 + kbB;

  v8f acc[10];
#pragma unroll
  for (int t = 0; t < 10; ++t)
#pragma unroll
    for (int j = 0; j < 8; ++j) acc[t][j] = 0.0f;

  for (int k0 = 0; k0 < KP3; k0 += 32) {
    // uniform (scalar) segment select: k0 is the loop counter
    const _Float16* p = (k0 < 1024) ? (a0 + k0)
                      : (k0 < 2048) ? (a1 + (k0 - 1024))
                      : (k0 < 2560) ? (a2 + (k0 - 2048))
                                    : a3;
    v16h a = pack2(*(const h8*)p, *(const h8*)(p + 16));
#pragma unroll
    for (int g = 0; g < 2; ++g) {
      v16h b0, b1, b2, b3, b4;
      {
        const _Float16* bp0 = brow + (size_t)(g * 5 + 0) * 16 * KP3 + k0;
        const _Float16* bp1 = brow + (size_t)(g * 5 + 1) * 16 * KP3 + k0;
        const _Float16* bp2 = brow + (size_t)(g * 5 + 2) * 16 * KP3 + k0;
        const _Float16* bp3 = brow + (size_t)(g * 5 + 3) * 16 * KP3 + k0;
        const _Float16* bp4 = brow + (size_t)(g * 5 + 4) * 16 * KP3 + k0;
        b0 = pack2(*(const h8*)bp0, *(const h8*)(bp0 + 16));
        b1 = pack2(*(const h8*)bp1, *(const h8*)(bp1 + 16));
        b2 = pack2(*(const h8*)bp2, *(const h8*)(bp2 + 16));
        b3 = pack2(*(const h8*)bp3, *(const h8*)(bp3 + 16));
        b4 = pack2(*(const h8*)bp4, *(const h8*)(bp4 + 16));
      }
      SCHED_FENCE();  // keep the 10 loads grouped ahead of the 5 WMMAs
      acc[g * 5 + 0] = __builtin_amdgcn_wmma_f32_16x16x32_f16(
          false, a, false, b0, (short)0, acc[g * 5 + 0], false, false);
      acc[g * 5 + 1] = __builtin_amdgcn_wmma_f32_16x16x32_f16(
          false, a, false, b1, (short)0, acc[g * 5 + 1], false, false);
      acc[g * 5 + 2] = __builtin_amdgcn_wmma_f32_16x16x32_f16(
          false, a, false, b2, (short)0, acc[g * 5 + 2], false, false);
      acc[g * 5 + 3] = __builtin_amdgcn_wmma_f32_16x16x32_f16(
          false, a, false, b3, (short)0, acc[g * 5 + 3], false, false);
      acc[g * 5 + 4] = __builtin_amdgcn_wmma_f32_16x16x32_f16(
          false, a, false, b4, (short)0, acc[g * 5 + 4], false, false);
    }
  }

  const int rowoff = half * 8;
#pragma unroll
  for (int t = 0; t < 10; ++t) {
    const int n = t * 16 + l16;
    const float bv = (n < 150) ? bias[n] : 0.0f;
#pragma unroll
    for (int j = 0; j < 8; ++j) {
      float v = acc[t][j] + bv;
      v = v > 0.0f ? v : 0.0f;
      Y[(size_t)(m0 + rowoff + j) * NPAD + n] = (_Float16)v;
    }
  }
}

// ----------------------- span softmax pooling ------------------------------
// One wave per span; each lane owns 4 contiguous embed dims x 4 chunks.
// Also emits the padded f16 width-embedding row sizeH[s][0..31].
__global__ void __launch_bounds__(256)
k_pool(const float* __restrict__ attns,
       const float* __restrict__ embeds,
       const int* __restrict__ starts,
       const int* __restrict__ lens,
       const float* __restrict__ width_table,
       _Float16* __restrict__ pooledH,
       _Float16* __restrict__ sizeH) {
  const int lane = threadIdx.x & 31;
  const int wid  = threadIdx.x >> 5;
  const int s    = blockIdx.x * 8 + wid;
  const int start = starts[s];
  const int len   = lens[s];

  // width bucket: searchsorted(BINS, len+1, 'right')
  const int width = len + 1;
  const int bins[8] = {1, 2, 3, 4, 8, 16, 32, 64};
  int bucket = 0;
#pragma unroll
  for (int i = 0; i < 8; ++i) bucket += (bins[i] <= width) ? 1 : 0;
  sizeH[(size_t)s * 32 + lane] =
      (lane < 20) ? (_Float16)width_table[bucket * 20 + lane] : (_Float16)0.0f;

  float w[10];
  float mx = -3.4e38f;
#pragma unroll
  for (int l = 0; l < 10; ++l) {
    const float v = (l <= len) ? attns[start + l] : -3.4e38f;
    w[l] = v;
    mx = fmaxf(mx, v);
  }
  float sum = 0.0f;
#pragma unroll
  for (int l = 0; l < 10; ++l) {
    const float e = (l <= len) ? __expf(w[l] - mx) : 0.0f;
    w[l] = e;
    sum += e;
  }
  const float inv = 1.0f / sum;

#pragma unroll
  for (int c = 0; c < 4; ++c) {
    const int col = c * 128 + lane * 4;
    f4 acc = {0.0f, 0.0f, 0.0f, 0.0f};
#pragma unroll
    for (int l = 0; l < 10; ++l) {
      if (l <= len) {
        f4 e = *(const f4*)(embeds + (size_t)(start + l) * E_DIM + col);
        acc += e * w[l];
      }
    }
    acc *= inv;
    h4 o;
    o[0] = (_Float16)acc[0]; o[1] = (_Float16)acc[1];
    o[2] = (_Float16)acc[2]; o[3] = (_Float16)acc[3];
    *(h4*)(pooledH + (size_t)s * E_DIM + col) = o;
  }
}

// ----------------------- final 150-dim dot layer ---------------------------
// out[m] = b + sum_{j<150} Xh[m,j] * w[j]. One wave per row.
__global__ void __launch_bounds__(256)
k_dot150(const _Float16* __restrict__ Xh,
         const float* __restrict__ w,
         const float* __restrict__ b,
         float* __restrict__ out) {
  const int lane = threadIdx.x & 31;
  const int m    = blockIdx.x * 8 + (threadIdx.x >> 5);
  const _Float16* row = Xh + (size_t)m * NPAD;
  float sum = 0.0f;
  for (int j = lane; j < 150; j += 32) sum += (float)row[j] * w[j];
  for (int off = 16; off > 0; off >>= 1) sum += __shfl_xor(sum, off, 32);
  if (lane == 0) out[m] = sum + b[0];
}

// ---------------------------------------------------------------------------

extern "C" void kernel_launch(void* const* d_in, const int* in_sizes, int n_in,
                              void* d_out, int out_size, void* d_ws, size_t ws_size,
                              hipStream_t stream) {
  (void)in_sizes; (void)n_in; (void)out_size; (void)ws_size;

  const float* states      = (const float*)d_in[0];
  const float* embeds      = (const float*)d_in[1];
  const int*   starts      = (const int*)d_in[2];
  const int*   lens        = (const int*)d_in[3];
  const float* Wa1         = (const float*)d_in[4];
  const float* ba1         = (const float*)d_in[5];
  const float* Wa2         = (const float*)d_in[6];
  const float* ba2         = (const float*)d_in[7];
  const float* Wa3         = (const float*)d_in[8];
  const float* ba3         = (const float*)d_in[9];
  const float* width_table = (const float*)d_in[10];
  const float* Ws1         = (const float*)d_in[11];
  const float* bs1         = (const float*)d_in[12];
  const float* Ws2         = (const float*)d_in[13];
  const float* bs2         = (const float*)d_in[14];
  const float* Ws3         = (const float*)d_in[15];
  const float* bs3         = (const float*)d_in[16];
  float* out = (float*)d_out;

  char* ws = (char*)d_ws;
  size_t off = 0;
  auto carve = [&](size_t bytes) -> char* {
    char* p = ws + off;
    off = (off + bytes + 255) & ~(size_t)255;
    return p;
  };
  _Float16* statesH = (_Float16*)carve((size_t)T_TOK * A_DIM * 2); // 16 MB
  _Float16* Wa1T    = (_Float16*)carve((size_t)NPAD * 1024 * 2);
  _Float16* Wa2T    = (_Float16*)carve((size_t)NPAD * NPAD * 2);
  _Float16* Ws1T    = (_Float16*)carve((size_t)NPAD * KP3 * 2);
  _Float16* Ws2T    = (_Float16*)carve((size_t)NPAD * NPAD * 2);
  _Float16* H1      = (_Float16*)carve((size_t)T_TOK * NPAD * 2);
  _Float16* H2      = (_Float16*)carve((size_t)T_TOK * NPAD * 2);
  float*    attns   = (float*)carve((size_t)T_TOK * 4);
  _Float16* pooledH = (_Float16*)carve((size_t)S_SPN * E_DIM * 2); // 33.5 MB
  _Float16* sizeH   = (_Float16*)carve((size_t)S_SPN * 32 * 2);    // 2 MB
  _Float16* S1      = (_Float16*)carve((size_t)S_SPN * NPAD * 2);
  _Float16* S2      = (_Float16*)carve((size_t)S_SPN * NPAD * 2);

  // 1) states -> f16
  k_cvt<<<512, 256, 0, stream>>>(states, statesH, T_TOK * A_DIM);
  // 2) weights -> transposed/padded f16
  k_prep_weights<<<1024, 256, 0, stream>>>(Wa1, Wa2, Ws1, Ws2, Wa1T, Wa2T, Ws1T, Ws2T);
  // 3-5) token attention MLP
  k_gemm_relu<<<T_TOK / 64, 128, 0, stream>>>(statesH, 1024, Wa1T, ba1, H1);
  k_gemm_relu<<<T_TOK / 64, 128, 0, stream>>>(H1, NPAD, Wa2T, ba2, H2);
  k_dot150<<<T_TOK / 8, 256, 0, stream>>>(H2, Wa3, ba3, attns);
  // 6) span softmax pooling + width embedding row
  k_pool<<<S_SPN / 8, 256, 0, stream>>>(attns, embeds, starts, lens,
                                        width_table, pooledH, sizeH);
  // 7-9) mention scoring MLP (fused gather GEMM + hidden GEMM + final dot)
  k_gemm_mention<<<S_SPN / 64, 128, 0, stream>>>(statesH, pooledH, sizeH,
                                                 starts, lens, Ws1T, bs1, S1);
  k_gemm_relu<<<S_SPN / 64, 128, 0, stream>>>(S1, NPAD, Ws2T, bs2, S2);
  k_dot150<<<S_SPN / 8, 256, 0, stream>>>(S2, Ws3, bs3, out);
}
